// FullAttention_38732015075820
// MI455X (gfx1250) — matrix operand order repaired and checked by reading, not compiled, not run on hardware
//
#include <hip/hip_runtime.h>
#include <math.h>
#include <stdint.h>

typedef __attribute__((ext_vector_type(16))) _Float16 v16h;
typedef __attribute__((ext_vector_type(8)))  _Float16 v8h;
typedef __attribute__((ext_vector_type(8)))  float    v8f;

union V16 { v16h v; v8h h[2]; };
union H16 { _Float16 a[16]; v8h h[2]; };

#define WMMA_F32_F16(a, b, c) \
  __builtin_amdgcn_wmma_f32_16x16x32_f16(false, (a), false, (b), (short)0, (c), false, false)

constexpr int B_ = 2, L_ = 2048, H_ = 16, E_ = 64;
constexpr int BM = 64;    // query rows per block (4 waves x 16)
constexpr int BN = 32;    // keys per kv tile
constexpr int KSTR = 72;  // padded halves per key row in LDS  ([key][e])
constexpr int VSTR = 40;  // padded halves per e row in LDS    ([e][key])
constexpr int PSTR = 40;  // padded halves per row of per-wave P staging

// ---------------- DPP 16-lane reductions (replaces ds_bpermute shuffles) ----------
template <int CTRL>
__device__ __forceinline__ float dpp_f(float x) {
  int r = __builtin_amdgcn_update_dpp(0, __builtin_bit_cast(int, x), CTRL, 0xf, 0xf, true);
  return __builtin_bit_cast(float, r);
}
__device__ __forceinline__ float rowmax16(float x) {
  x = fmaxf(x, dpp_f<0xB1>(x));   // quad_perm [1,0,3,2]  (xor 1)
  x = fmaxf(x, dpp_f<0x4E>(x));   // quad_perm [2,3,0,1]  (xor 2)
  x = fmaxf(x, dpp_f<0x141>(x));  // row_half_mirror      (xor 4 after quads uniform)
  x = fmaxf(x, dpp_f<0x140>(x));  // row_mirror           (xor 8)
  return x;
}
__device__ __forceinline__ float rowsum16(float x) {
  x += dpp_f<0xB1>(x);
  x += dpp_f<0x4E>(x);
  x += dpp_f<0x141>(x);
  x += dpp_f<0x140>(x);
  return x;
}

// ---------------- one-time f32 -> f16 convert: K -> [bh][s][e], V -> [bh][e][s] ----
__global__ __launch_bounds__(256)
void cvt_kv(const float* __restrict__ K, const float* __restrict__ V,
            _Float16* __restrict__ Kc, _Float16* __restrict__ Vt)
{
  __shared__ _Float16 Vl[64 * 72];
  const int tid = threadIdx.x;
  const int blk = blockIdx.x;
  const int st = blk & (L_ / 64 - 1);
  const int hd = (blk >> 5) & (H_ - 1);
  const int b  = blk >> 9;
  const int s0 = st * 64;
  const int bh = b * H_ + hd;
  const int row = tid >> 2;          // 0..63
  const int c4  = (tid & 3) * 16;    // e / key chunk base

  { // K: straight convert, layout [bh][s][e]
    const float* kp = K + (((size_t)b * L_ + s0 + row) * H_ + hd) * E_ + c4;
    H16 t;
    #pragma unroll
    for (int i = 0; i < 16; i += 4) {
      float4 f = *(const float4*)(kp + i);
      t.a[i] = (_Float16)f.x; t.a[i+1] = (_Float16)f.y;
      t.a[i+2] = (_Float16)f.z; t.a[i+3] = (_Float16)f.w;
    }
    _Float16* kd = Kc + ((size_t)bh * L_ + s0 + row) * E_ + c4;
    *(v8h*)(kd) = t.h[0];  *(v8h*)(kd + 8) = t.h[1];
  }
  { // V: LDS-tiled transpose to [bh][e][s]
    const float* vp = V + (((size_t)b * L_ + s0 + row) * H_ + hd) * E_ + c4;
    #pragma unroll
    for (int i = 0; i < 16; i += 4) {
      float4 f = *(const float4*)(vp + i);
      Vl[row * 72 + c4 + i + 0] = (_Float16)f.x;
      Vl[row * 72 + c4 + i + 1] = (_Float16)f.y;
      Vl[row * 72 + c4 + i + 2] = (_Float16)f.z;
      Vl[row * 72 + c4 + i + 3] = (_Float16)f.w;
    }
    __syncthreads();
    const int e = row, sc = c4;
    H16 t;
    #pragma unroll
    for (int i = 0; i < 16; ++i) t.a[i] = Vl[(sc + i) * 72 + e];
    _Float16* vd = Vt + ((size_t)bh * E_ + e) * L_ + s0 + sc;
    *(v8h*)(vd) = t.h[0];  *(v8h*)(vd + 8) = t.h[1];
  }
}

// ---------------- async DMA staging of one kv tile into LDS (f16, pre-laid-out) ---
__device__ __forceinline__ void stage_async(const _Float16* __restrict__ Kc_bh,
                                            const _Float16* __restrict__ Vt_bh,
                                            _Float16* KtBuf, _Float16* VtBuf,
                                            int s0, int tid)
{
  #pragma unroll
  for (int j = 0; j < 2; ++j) {          // K tile: 32 rows x 128B
    const int idx = tid * 2 + j;
    const int row = idx >> 3, ch = idx & 7;
    uint64_t g = (uint64_t)(uintptr_t)(Kc_bh + (size_t)(s0 + row) * E_) + (uint64_t)ch * 16;
    uint32_t l = (uint32_t)(uintptr_t)(KtBuf + row * KSTR) + (uint32_t)ch * 16;
    asm volatile("global_load_async_to_lds_b128 %0, %1, off"
                 :: "v"(l), "v"(g) : "memory");
  }
  #pragma unroll
  for (int j = 0; j < 2; ++j) {          // V tile (transposed): 64 e-rows x 64B
    const int idx = tid * 2 + j;
    const int e = idx >> 2, ch = idx & 3;
    uint64_t g = (uint64_t)(uintptr_t)(Vt_bh + (size_t)e * L_ + s0) + (uint64_t)ch * 16;
    uint32_t l = (uint32_t)(uintptr_t)(VtBuf + e * VSTR) + (uint32_t)ch * 16;
    asm volatile("global_load_async_to_lds_b128 %0, %1, off"
                 :: "v"(l), "v"(g) : "memory");
  }
}

// ---------------- per-tile compute: S = Q K^T, online softmax, O += P V ----------
__device__ __forceinline__ void attend_tile(const _Float16* __restrict__ Ktb,
                                            const _Float16* __restrict__ Vtb,
                                            _Float16* __restrict__ Ptw,
                                            const V16 qa[2],
                                            float mrow[8], float lrow[8], v8f of[4],
                                            int s0, int m0w, int col, int hgrp)
{
  const float c2 = 0.125f * 1.44269504088896340736f;  // 1/sqrt(E) * log2(e)

  V16 kb[2][2];
  #pragma unroll
  for (int nh = 0; nh < 2; ++nh)
    #pragma unroll
    for (int c = 0; c < 2; ++c) {
      const _Float16* p = &Ktb[(nh * 16 + col) * KSTR + c * 32 + hgrp * 16];
      kb[nh][c].h[0] = *(const v8h*)(p);
      kb[nh][c].h[1] = *(const v8h*)(p + 8);
    }

  v8f sA = {}, sB = {};
  sA = WMMA_F32_F16(qa[0].v, kb[0][0].v, sA);   // keys s0 .. s0+15
  sA = WMMA_F32_F16(qa[1].v, kb[0][1].v, sA);
  sB = WMMA_F32_F16(qa[0].v, kb[1][0].v, sB);   // keys s0+16 .. s0+31
  sB = WMMA_F32_F16(qa[1].v, kb[1][1].v, sB);

  const int k0 = s0 + col, k1 = s0 + 16 + col;
  float alpha[8];
  #pragma unroll
  for (int r = 0; r < 8; ++r) {
    const int row = m0w + hgrp * 8 + r;
    float z0 = (k0 <= row) ? sA[r] * c2 : -INFINITY;
    float z1 = (k1 <= row) ? sB[r] * c2 : -INFINITY;
    float tm = rowmax16(fmaxf(z0, z1));
    const float mold = mrow[r];
    const float mnew = fmaxf(mold, tm);
    const float a  = exp2f(mold - mnew);
    const float p0 = exp2f(z0 - mnew);
    const float p1 = exp2f(z1 - mnew);
    const float rs = rowsum16(p0 + p1);
    mrow[r]  = mnew;
    lrow[r]  = lrow[r] * a + rs;
    alpha[r] = a;
    Ptw[(hgrp * 8 + r) * PSTR + col]      = (_Float16)p0;  // park P (C-layout)
    Ptw[(hgrp * 8 + r) * PSTR + col + 16] = (_Float16)p1;
  }

  asm volatile("s_wait_dscnt 0x0" ::: "memory");  // same-wave LDS transpose fence

  V16 pa;  // P re-read as 16x32 A-fragment
  pa.h[0] = *(const v8h*)&Ptw[col * PSTR + hgrp * 16];
  pa.h[1] = *(const v8h*)&Ptw[col * PSTR + hgrp * 16 + 8];

  #pragma unroll
  for (int ec = 0; ec < 4; ++ec) {
    V16 vb;  // V B-fragment: contiguous keys at fixed e
    const _Float16* p = &Vtb[(ec * 16 + col) * VSTR + hgrp * 16];
    vb.h[0] = *(const v8h*)(p);
    vb.h[1] = *(const v8h*)(p + 8);
    v8f o = of[ec];
    #pragma unroll
    for (int r = 0; r < 8; ++r) o[r] *= alpha[r];
    of[ec] = WMMA_F32_F16(pa.v, vb.v, o);
  }
}

// ---------------- main attention kernel --------------------------------------------
template <bool ASYNC>
__global__ __launch_bounds__(128)
void fa_fwd_causal(const float* __restrict__ Q, const float* __restrict__ K,
                   const float* __restrict__ V,
                   const _Float16* __restrict__ Kc, const _Float16* __restrict__ Vt,
                   float* __restrict__ O)
{
  __shared__ _Float16 Kt[2][BN * KSTR];    // double-buffered [key][e]
  __shared__ _Float16 VtL[2][E_ * VSTR];   // double-buffered [e][key]
  __shared__ _Float16 Pt[4][16 * PSTR];

  const int tid  = threadIdx.x;
  const int lane = tid & 31;
  const int wave = tid >> 5;
  const int col  = lane & 15;
  const int hgrp = lane >> 4;

  const int blk = blockIdx.x;
  const int mt  = blk & (L_ / BM - 1);
  const int hd  = (blk >> 5) & (H_ - 1);
  const int b   = blk >> 9;
  const int m0  = mt * BM;
  const int m0w = m0 + wave * 16;
  const int bh  = b * H_ + hd;

  // Q -> two 16x32 f16 A-fragments, held in registers
  V16 qa[2];
  {
    const float* qb = Q + (((size_t)b * L_ + m0w + col) * H_ + hd) * E_;
    #pragma unroll
    for (int c = 0; c < 2; ++c) {
      const int e0 = c * 32 + hgrp * 16;
      #pragma unroll
      for (int i = 0; i < 8; ++i) {
        qa[c].v[i]     = (_Float16)qb[e0 + i];
        qa[c].v[i + 8] = (_Float16)qb[e0 + 8 + i];
      }
    }
  }

  float mrow[8], lrow[8];
  v8f   of[4] = {};
  #pragma unroll
  for (int r = 0; r < 8; ++r) { mrow[r] = -INFINITY; lrow[r] = 0.0f; }

  const int ntiles = m0 / BN + 2;   // causal: tiles with s0 <= m0+63

  const _Float16* Kc_bh = Kc + (size_t)bh * L_ * E_;
  const _Float16* Vt_bh = Vt + (size_t)bh * E_ * L_;

  if constexpr (ASYNC) {
    stage_async(Kc_bh, Vt_bh, Kt[0], VtL[0], 0, tid);   // prologue DMA of tile 0
    for (int t = 0; t < ntiles; ++t) {
      asm volatile("s_wait_asynccnt 0x0" ::: "memory"); // my transfers landed
      __syncthreads();                                  // everyone's landed; prev compute done
      if (t + 1 < ntiles)                               // overlap next DMA with compute
        stage_async(Kc_bh, Vt_bh, Kt[(t + 1) & 1], VtL[(t + 1) & 1], (t + 1) * BN, tid);
      if (t * BN <= m0w + 15)
        attend_tile(Kt[t & 1], VtL[t & 1], Pt[wave], qa, mrow, lrow, of,
                    t * BN, m0w, col, hgrp);
    }
  } else {
    // fallback: in-kernel f32->f16 convert staging (single buffer)
    const int lr = tid >> 4;
    const int lc = (tid & 15) * 4;
    for (int t = 0; t < ntiles; ++t) {
      const int s0 = t * BN;
      __syncthreads();
      #pragma unroll
      for (int rr = 0; rr < BN; rr += 8) {
        const int row = rr + lr;
        const float* kp = K + (((size_t)b * L_ + s0 + row) * H_ + hd) * E_ + lc;
        const float* vp = V + (((size_t)b * L_ + s0 + row) * H_ + hd) * E_ + lc;
        const float4 k4 = *(const float4*)kp;
        const float4 v4 = *(const float4*)vp;
        Kt[0][row * KSTR + lc + 0] = (_Float16)k4.x;
        Kt[0][row * KSTR + lc + 1] = (_Float16)k4.y;
        Kt[0][row * KSTR + lc + 2] = (_Float16)k4.z;
        Kt[0][row * KSTR + lc + 3] = (_Float16)k4.w;
        VtL[0][(lc + 0) * VSTR + row] = (_Float16)v4.x;
        VtL[0][(lc + 1) * VSTR + row] = (_Float16)v4.y;
        VtL[0][(lc + 2) * VSTR + row] = (_Float16)v4.z;
        VtL[0][(lc + 3) * VSTR + row] = (_Float16)v4.w;
        if (t + 1 < ntiles) {
          __builtin_prefetch(kp + (size_t)BN * H_ * E_, 0, 0);
          __builtin_prefetch(vp + (size_t)BN * H_ * E_, 0, 0);
        }
      }
      __syncthreads();
      if (s0 <= m0w + 15)
        attend_tile(Kt[0], VtL[0], Pt[wave], qa, mrow, lrow, of, s0, m0w, col, hgrp);
    }
  }

  // epilogue: normalize and store
  #pragma unroll
  for (int r = 0; r < 8; ++r) {
    const int row = m0w + hgrp * 8 + r;
    const float inv = 1.0f / lrow[r];
    float* ob = O + (((size_t)b * L_ + row) * H_ + hd) * E_;
    #pragma unroll
    for (int ec = 0; ec < 4; ++ec)
      ob[ec * 16 + col] = of[ec][r] * inv;
  }
}

extern "C" void kernel_launch(void* const* d_in, const int* in_sizes, int n_in,
                              void* d_out, int out_size, void* d_ws, size_t ws_size,
                              hipStream_t stream) {
  (void)in_sizes; (void)n_in; (void)out_size;
  const float* q = (const float*)d_in[0];
  const float* k = (const float*)d_in[1];
  const float* v = (const float*)d_in[2];
  float* o = (float*)d_out;

  const size_t elems = (size_t)B_ * H_ * L_ * E_;
  const size_t need  = elems * 2 /*bytes*/ * 2 /*K + V*/;
  dim3 grid(B_ * H_ * (L_ / BM));
  dim3 block(128);

  if (ws_size >= need) {
    _Float16* Kc = (_Float16*)d_ws;
    _Float16* Vt = Kc + elems;
    hipLaunchKernelGGL(cvt_kv, dim3(B_ * H_ * (L_ / 64)), dim3(256), 0, stream, k, v, Kc, Vt);
    hipLaunchKernelGGL((fa_fwd_causal<true>), grid, block, 0, stream, q, k, v, Kc, Vt, o);
  } else {
    hipLaunchKernelGGL((fa_fwd_causal<false>), grid, block, 0, stream,
                       q, k, v, (const _Float16*)nullptr, (const _Float16*)nullptr, o);
  }
}
